// LSTM_49649821942476
// MI455X (gfx1250) — matrix operand order repaired and checked by reading, not compiled, not run on hardware
//
#include <hip/hip_runtime.h>

// ---------------- problem constants ----------------
#define H   51
#define B   256
#define T   2048
#define FUT 64
#define TT  (T + FUT)

#define NPAD 256   // 4 gates x 64 padded hidden
#define K0   64    // layer0 K (h0 padded)
#define K1   128   // layer1 K (h0|h1, each padded to 64)

// LDS layout (bytes)
#define SX_ELEMS   (16 * T)            // staged x slice (f32)
#define WT0_ELEMS  (NPAD * K0)         // f16
#define WT1_ELEMS  (NPAD * K1)         // f16
#define SMEM_BYTES (SX_ELEMS*4 + WT0_ELEMS*2 + WT1_ELEMS*2 + 3*NPAD*4 + 16*K1*2 + 16*4 + 64*4)

typedef __attribute__((ext_vector_type(16))) _Float16 v16h;
typedef __attribute__((ext_vector_type(8)))  _Float16 v8h;
typedef __attribute__((ext_vector_type(8)))  float    v8f;

static __device__ __forceinline__ v8f wmma16x16x32(v16h a, v16h b, v8f c) {
    // (neg_a, A, neg_b, B, c_mod, C, reuse_a, reuse_b)
    return __builtin_amdgcn_wmma_f32_16x16x32_f16(false, a, false, b, (short)0, c, false, false);
}

static __device__ __forceinline__ float sigf(float v) {
    return __builtin_amdgcn_rcpf(1.0f + __expf(-v));
}
static __device__ __forceinline__ float tanhfast(float v) {
    float e = __expf(2.0f * v);
    return (e - 1.0f) * __builtin_amdgcn_rcpf(e + 1.0f);
}

// A-fragment (16x32 f16): two 16B-contiguous 8-half chunks per lane
static __device__ __forceinline__ v16h load_a(const _Float16* p0, const _Float16* p1) {
    v8h lo = *(const v8h*)p0;
    v8h hi = *(const v8h*)p1;
    v16h a;
#pragma unroll
    for (int i = 0; i < 8; ++i) { a[i] = lo[i]; a[i + 8] = hi[i]; }
    return a;
}

__global__ __launch_bounds__(128) void lstm_wmma_kernel(
    const float* __restrict__ x,
    const float* __restrict__ W_ih0, const float* __restrict__ W_hh0,
    const float* __restrict__ b_ih0, const float* __restrict__ b_hh0,
    const float* __restrict__ W_ih1, const float* __restrict__ W_hh1,
    const float* __restrict__ b_ih1, const float* __restrict__ b_hh1,
    const float* __restrict__ fc_w, const float* __restrict__ fc_b,
    float* __restrict__ out)
{
    extern __shared__ unsigned char smem[];
    float*    sx    = (float*)smem;                      // [16][T]
    _Float16* sWt0  = (_Float16*)(sx + SX_ELEMS);        // [NPAD][K0]  (B layout: [n][k])
    _Float16* sWt1  = sWt0 + WT0_ELEMS;                  // [NPAD][K1]
    float*    sb0   = (float*)(sWt1 + WT1_ELEMS);        // [NPAD]
    float*    sb1   = sb0 + NPAD;                        // [NPAD]
    float*    swx0  = sb1 + NPAD;                        // [NPAD] W_ih0 column
    _Float16* hcat  = (_Float16*)(swx0 + NPAD);          // [16][K1] : h0 | h1 (f16, zero padded)
    float*    sxt   = (float*)(hcat + 16 * K1);          // [16] current step input
    float*    sypart= sxt + 16;                          // [4][16] per-wave head partials

    const int tid    = threadIdx.x;
    const int wgbase = blockIdx.x * 16;   // 16 batch rows per workgroup

    // ---------------- one-time prep ----------------
    // stage x slice (coalesced)
    for (int idx = tid; idx < SX_ELEMS; idx += 128) {
        int m = idx >> 11, tc = idx & (T - 1);
        sx[m * T + tc] = x[(wgbase + m) * T + tc];
    }
    // layer0 recurrent weights, padded: row np = G*64 + j, col k
    for (int idx = tid; idx < WT0_ELEMS; idx += 128) {
        int np = idx / K0, k = idx % K0;
        int G = np >> 6, j = np & 63;
        float v = 0.0f;
        if (j < H && k < H) v = W_hh0[(G * H + j) * H + k];
        sWt0[idx] = (_Float16)v;
    }
    // layer1 concat weights: k<64 -> W_ih1 (vs h0), k>=64 -> W_hh1 (vs h1)
    for (int idx = tid; idx < WT1_ELEMS; idx += 128) {
        int np = idx / K1, k = idx % K1;
        int G = np >> 6, j = np & 63;
        float v = 0.0f;
        if (j < H) {
            if (k < H)                       v = W_ih1[(G * H + j) * H + k];
            else if (k >= 64 && k < 64 + H)  v = W_hh1[(G * H + j) * H + (k - 64)];
        }
        sWt1[idx] = (_Float16)v;
    }
    for (int idx = tid; idx < NPAD; idx += 128) {
        int G = idx >> 6, j = idx & 63;
        float v0 = 0.f, v1 = 0.f, w0 = 0.f;
        if (j < H) {
            int nsrc = G * H + j;
            v0 = b_ih0[nsrc] + b_hh0[nsrc];
            v1 = b_ih1[nsrc] + b_hh1[nsrc];
            w0 = W_ih0[nsrc];
        }
        sb0[idx] = v0; sb1[idx] = v1; swx0[idx] = w0;
    }
    for (int idx = tid; idx < 16 * K1; idx += 128) hcat[idx] = (_Float16)0.0f;
    if (tid < 16) sxt[tid] = x[(wgbase + tid) * T];
    __syncthreads();

    // ---------------- per-lane constants ----------------
    const int lane    = tid & 31;
    const int wave    = tid >> 5;        // 0..3 -> hidden chunk
    const int n       = lane & 15;       // column within 16x16 tile
    const int halfsel = lane >> 4;       // C layout: rows m = halfsel*8 + r
    const int mbase   = halfsel * 8;
    const int j       = wave * 16 + n;   // hidden unit this lane produces

    // hoist all B fragments (loop-invariant weights) into VGPRs
    v16h B0[4][2], B1[4][4];
    float bi0[4], wx0[4], bi1[4];
#pragma unroll
    for (int G = 0; G < 4; ++G) {
        const int nprow = G * 64 + wave * 16 + n;   // B-frag: lane holds column n of the tile
#pragma unroll
        for (int kc = 0; kc < 2; ++kc)
            B0[G][kc] = load_a(sWt0 + nprow * K0 + kc * 32 + halfsel * 8, sWt0 + nprow * K0 + kc * 32 + 16 + halfsel * 8);
#pragma unroll
        for (int kc = 0; kc < 4; ++kc)
            B1[G][kc] = load_a(sWt1 + nprow * K1 + kc * 32 + halfsel * 8, sWt1 + nprow * K1 + kc * 32 + 16 + halfsel * 8);
        bi0[G] = sb0[nprow]; wx0[G] = swx0[nprow]; bi1[G] = sb1[nprow];
    }
    const float fcw = (j < H) ? fc_w[j] : 0.0f;
    const float fcb = fc_b[0];

    v8f c0 = {};   // cell state, WMMA C layout: (m = mbase+r, hidden j)
    v8f c1 = {};
    const _Float16* arow = hcat + n * K1;   // A-frag: lane's matrix row m == n

    // ---------------- 2112 sequential timesteps ----------------
    for (int t = 0; t < TT; ++t) {
        // ---- layer 0 ----
        float xv[8];
#pragma unroll
        for (int r = 0; r < 8; ++r) xv[r] = sxt[mbase + r];
        v16h a00 = load_a(arow + 0  + halfsel * 8, arow + 16 + halfsel * 8);
        v16h a01 = load_a(arow + 32 + halfsel * 8, arow + 48 + halfsel * 8);
        __syncthreads();   // all reads of h0_old / sxt complete

        v8f gi, gf, gg, go;
#pragma unroll
        for (int r = 0; r < 8; ++r) {
            gi[r] = bi0[0] + xv[r] * wx0[0];
            gf[r] = bi0[1] + xv[r] * wx0[1];
            gg[r] = bi0[2] + xv[r] * wx0[2];
            go[r] = bi0[3] + xv[r] * wx0[3];
        }
        gi = wmma16x16x32(a00, B0[0][0], gi); gi = wmma16x16x32(a01, B0[0][1], gi);
        gf = wmma16x16x32(a00, B0[1][0], gf); gf = wmma16x16x32(a01, B0[1][1], gf);
        gg = wmma16x16x32(a00, B0[2][0], gg); gg = wmma16x16x32(a01, B0[2][1], gg);
        go = wmma16x16x32(a00, B0[3][0], go); go = wmma16x16x32(a01, B0[3][1], go);

        v8f h0v;
#pragma unroll
        for (int r = 0; r < 8; ++r) {
            float c = sigf(gf[r]) * c0[r] + sigf(gi[r]) * tanhfast(gg[r]);
            c0[r] = c;
            h0v[r] = sigf(go[r]) * tanhfast(c);
        }
        if (j < H) {
#pragma unroll
            for (int r = 0; r < 8; ++r)
                hcat[(mbase + r) * K1 + j] = (_Float16)h0v[r];
        }
        __syncthreads();   // h0_new visible

        // ---- layer 1 (K = h0_new | h1_old) ----
        v16h a10 = load_a(arow + 0   + halfsel * 8, arow + 16  + halfsel * 8);
        v16h a11 = load_a(arow + 32  + halfsel * 8, arow + 48  + halfsel * 8);
        v16h a12 = load_a(arow + 64  + halfsel * 8, arow + 80  + halfsel * 8);
        v16h a13 = load_a(arow + 96  + halfsel * 8, arow + 112 + halfsel * 8);
        __syncthreads();   // all reads of h1_old complete

#pragma unroll
        for (int r = 0; r < 8; ++r) { gi[r] = bi1[0]; gf[r] = bi1[1]; gg[r] = bi1[2]; go[r] = bi1[3]; }
        gi = wmma16x16x32(a10, B1[0][0], gi); gi = wmma16x16x32(a11, B1[0][1], gi);
        gi = wmma16x16x32(a12, B1[0][2], gi); gi = wmma16x16x32(a13, B1[0][3], gi);
        gf = wmma16x16x32(a10, B1[1][0], gf); gf = wmma16x16x32(a11, B1[1][1], gf);
        gf = wmma16x16x32(a12, B1[1][2], gf); gf = wmma16x16x32(a13, B1[1][3], gf);
        gg = wmma16x16x32(a10, B1[2][0], gg); gg = wmma16x16x32(a11, B1[2][1], gg);
        gg = wmma16x16x32(a12, B1[2][2], gg); gg = wmma16x16x32(a13, B1[2][3], gg);
        go = wmma16x16x32(a10, B1[3][0], go); go = wmma16x16x32(a11, B1[3][1], go);
        go = wmma16x16x32(a12, B1[3][2], go); go = wmma16x16x32(a13, B1[3][3], go);

        v8f h1v;
#pragma unroll
        for (int r = 0; r < 8; ++r) {
            float c = sigf(gf[r]) * c1[r] + sigf(gi[r]) * tanhfast(gg[r]);
            c1[r] = c;
            h1v[r] = sigf(go[r]) * tanhfast(c);
        }
        if (j < H) {
#pragma unroll
            for (int r = 0; r < 8; ++r)
                hcat[(mbase + r) * K1 + 64 + j] = (_Float16)h1v[r];
        }

        // ---- head: y = h1 . fc_w (+fc_b), reduce 16 lanes then 4 waves ----
        float pr[8];
#pragma unroll
        for (int r = 0; r < 8; ++r) pr[r] = h1v[r] * fcw;
#pragma unroll
        for (int off = 1; off < 16; off <<= 1) {
#pragma unroll
            for (int r = 0; r < 8; ++r) pr[r] += __shfl_xor(pr[r], off, 32);
        }
        if (n == 0) {
#pragma unroll
            for (int r = 0; r < 8; ++r) sypart[wave * 16 + mbase + r] = pr[r];
        }
        __syncthreads();

        if (tid < 16) {
            float y = sypart[tid] + sypart[16 + tid] + sypart[32 + tid] + sypart[48 + tid] + fcb;
            out[(wgbase + tid) * TT + t] = y;
            sxt[tid] = (t + 1 < T) ? sx[tid * T + (t + 1)] : y;  // autoregressive feed
        }
        __syncthreads();
    }
}

extern "C" void kernel_launch(void* const* d_in, const int* in_sizes, int n_in,
                              void* d_out, int out_size, void* d_ws, size_t ws_size,
                              hipStream_t stream) {
    (void)in_sizes; (void)n_in; (void)d_ws; (void)ws_size; (void)out_size;
    const float* x    = (const float*)d_in[0];
    const float* Wih0 = (const float*)d_in[1];
    const float* Whh0 = (const float*)d_in[2];
    const float* bih0 = (const float*)d_in[3];
    const float* bhh0 = (const float*)d_in[4];
    const float* Wih1 = (const float*)d_in[5];
    const float* Whh1 = (const float*)d_in[6];
    const float* bih1 = (const float*)d_in[7];
    const float* bhh1 = (const float*)d_in[8];
    const float* fcw  = (const float*)d_in[9];
    const float* fcb  = (const float*)d_in[10];
    float* out = (float*)d_out;

    (void)hipFuncSetAttribute((const void*)lstm_wmma_kernel,
                              hipFuncAttributeMaxDynamicSharedMemorySize, SMEM_BYTES);
    lstm_wmma_kernel<<<dim3(B / 16), dim3(128), SMEM_BYTES, stream>>>(
        x, Wih0, Whh0, bih0, bhh0, Wih1, Whh1, bih1, bhh1, fcw, fcb, out);
}
